// SparseGlobalAttention_72988674228668
// MI455X (gfx1250) — matrix operand order repaired and checked by reading, not compiled, not run on hardware
//
#include <hip/hip_runtime.h>
#include <cstdint>
#include <cstddef>

// ---------------- problem constants ----------------
constexpr int Bc  = 2;
constexpr int Tc  = 2048;
constexpr int Dc  = 768;
constexpr int Hc  = 12;
constexpr int HDc = 64;
constexpr int WINc = 16;
constexpr int Kc  = 2 * WINc + 1;   // 33 local keys
constexpr int Gc  = 32;             // global keys
constexpr int NKEY = Kc + Gc;       // 65
constexpr int Mc  = Bc * Tc;        // 4096 GEMM rows

// ---------------- WMMA / async types ----------------
typedef __attribute__((ext_vector_type(16))) __bf16   v16bf;
typedef __attribute__((ext_vector_type(8)))  float    v8f;
typedef __attribute__((ext_vector_type(8)))  uint32_t v8u;
typedef __attribute__((ext_vector_type(4)))  int      v4i;

#if __has_builtin(__builtin_amdgcn_global_load_async_to_lds_b128)
#define HAVE_ASYNC_LDS 1
#else
#define HAVE_ASYNC_LDS 0
#endif

#define AS_GLOBAL __attribute__((address_space(1)))
#define AS_LDS    __attribute__((address_space(3)))

__device__ __forceinline__ void wait_async_all() {
#if __has_builtin(__builtin_amdgcn_s_wait_asynccnt)
    __builtin_amdgcn_s_wait_asynccnt(0);
#elif defined(__AMDGCN__)
    asm volatile("s_wait_asynccnt 0" ::: "memory");
#endif
}

// ---------------- helpers ----------------
__device__ __forceinline__ uint16_t f2bf(float f) {
    uint32_t u = __float_as_uint(f);
    uint32_t r = (u + 0x7FFFu + ((u >> 16) & 1u)) >> 16;   // RNE
    return (uint16_t)r;
}

// ---------------- fp32 -> bf16 convert ----------------
__global__ __launch_bounds__(256) void f2bf_kernel(const float* __restrict__ in,
                                                   uint16_t* __restrict__ out, int n) {
    int i = blockIdx.x * 256 + threadIdx.x;
    if (i < n) out[i] = f2bf(in[i]);
}

// ---------------- fp32 W[K][N] -> bf16 W^T[N][K] ----------------
__global__ __launch_bounds__(256) void f2bf_transpose_kernel(const float* __restrict__ in,
                                                             uint16_t* __restrict__ out) {
    int i = blockIdx.x * 256 + threadIdx.x;     // over Dc*Dc
    if (i < Dc * Dc) {
        int n = i / Dc, k = i - n * Dc;
        out[i] = f2bf(in[(size_t)k * Dc + n]);  // coalesced writes, strided reads (tiny)
    }
}

// ---------------- gather global indices (ordered nonzero, padded with 0) ----------------
__global__ void build_gidx_kernel(const int* __restrict__ mask, int* __restrict__ gidx) {
    if (threadIdx.x == 0 && blockIdx.x == 0) {
        int c = 0;
        for (int t = 0; t < Tc && c < Gc; ++t)
            if (mask[t] != 0) gidx[c++] = t;
        for (; c < Gc; ++c) gidx[c] = 0;
    }
}

// ---------------- bf16 WMMA GEMM:  C[M,768] = A[M,768] * W[768,768] + bias ----------------
// W passed pre-transposed (WT[N][K]). Block tile 128x128, BK=32 (one wmma K step).
// 256 threads = 8 waves; wave w computes rows [16w,16w+16) x 128 cols -> 8 accumulators.
// A and B tiles staged into double-buffered LDS with async global->LDS copies.
__global__ __launch_bounds__(256) void gemm_bf16_kernel(const uint16_t* __restrict__ A,
                                                        const uint16_t* __restrict__ WT,
                                                        const float* __restrict__ bias,
                                                        float* __restrict__ C) {
    constexpr int BM = 128, BN = 128, BK = 32;
    constexpr int NT = Dc / BK;                       // 24 K-steps
    __shared__ __align__(16) uint16_t As[2][BM][BK];  // 16 KB
    __shared__ __align__(16) uint16_t Bs[2][BN][BK];  // 16 KB

    const int tid  = threadIdx.x;
    const int wave = tid >> 5;
    const int lane = tid & 31;
    const int half = lane >> 4;
    const int l16  = lane & 15;
    const int mBase = blockIdx.x * BM;
    const int nBase = blockIdx.y * BN;
    const int mOff  = wave * 16;

    v8f acc[8] = {};

    // stage one 128x32 bf16 tile (row stride Dc) into LDS buffer
    auto stage = [&](int buf, int kBase) {
        #pragma unroll
        for (int i = 0; i < 2; ++i) {
            int idx = tid + i * 256;          // 0..511
            int r   = idx >> 2;
            int seg = (idx & 3) * 8;
            const uint16_t* ga = A  + (size_t)(mBase + r) * Dc + kBase + seg;
            const uint16_t* gb = WT + (size_t)(nBase + r) * Dc + kBase + seg;
#if HAVE_ASYNC_LDS
            __builtin_amdgcn_global_load_async_to_lds_b128(
                (AS_GLOBAL v4i*)ga, (AS_LDS v4i*)&As[buf][r][seg], 0, 0);
            __builtin_amdgcn_global_load_async_to_lds_b128(
                (AS_GLOBAL v4i*)gb, (AS_LDS v4i*)&Bs[buf][r][seg], 0, 0);
#else
            *(uint4*)&As[buf][r][seg] = *(const uint4*)ga;
            *(uint4*)&Bs[buf][r][seg] = *(const uint4*)gb;
#endif
        }
    };

    stage(0, 0);

    for (int kt = 0; kt < NT; ++kt) {
        const int cur = kt & 1;
#if HAVE_ASYNC_LDS
        wait_async_all();                 // buf[cur] landed in LDS
#endif
        __syncthreads();                  // all waves see buf[cur]; done reading buf[1-cur]
        if (kt + 1 < NT) stage(1 - cur, (kt + 1) * BK);   // overlap with compute

        // A fragment (16x32 bf16): lanes 0-15 hold K {0..7,16..23}, lanes 16-31 {8..15,24..31}
        v8u au;
        const int kb = half * 8;
        #pragma unroll
        for (int i = 0; i < 4; ++i)
            au[i] = *(const uint32_t*)&As[cur][mOff + l16][kb + 2 * i];
        #pragma unroll
        for (int i = 0; i < 4; ++i)
            au[4 + i] = *(const uint32_t*)&As[cur][mOff + l16][16 + kb + 2 * i];
        v16bf af = __builtin_bit_cast(v16bf, au);

        #pragma unroll
        for (int s = 0; s < 8; ++s) {
            // B fragment (32x16 bf16): N = lane%16, lanes 0-15 K {0..7,16..23}, lanes 16-31 {8..15,24..31}
            v8u bu;
            const int kbb = half * 8;
            #pragma unroll
            for (int i = 0; i < 8; ++i)
                bu[i] = *(const uint32_t*)&Bs[cur][s * 16 + l16][(i >> 2) * 16 + kbb + 2 * (i & 3)];
            v16bf bf = __builtin_bit_cast(v16bf, bu);
            acc[s] = __builtin_amdgcn_wmma_f32_16x16x32_bf16(
                false, af, false, bf, (short)0, acc[s], false, false);
        }
    }

    // epilogue: D layout VGPR i -> M = i + 8*half, N = lane%16
    #pragma unroll
    for (int s = 0; s < 8; ++s) {
        const int ncol = nBase + s * 16 + l16;
        const float bval = bias[ncol];
        const int mrow0 = mBase + mOff + 8 * half;
        #pragma unroll
        for (int i = 0; i < 8; ++i)
            C[(size_t)(mrow0 + i) * Dc + ncol] = acc[s][i] + bval;
    }
}

// ---------------- sparse+global attention, one wave per (b,h,t) row ----------------
__global__ __launch_bounds__(256) void attn_kernel(const float* __restrict__ q,
                                                   const float* __restrict__ k,
                                                   const float* __restrict__ v,
                                                   const int*   __restrict__ mask,   // batch-0 row
                                                   const int*   __restrict__ gidx,
                                                   float* __restrict__ ctx,
                                                   float* __restrict__ full_attn) {
    __shared__ float qs[8][HDc];
    __shared__ float as[8][72];

    const int wave = threadIdx.x >> 5;
    const int lane = threadIdx.x & 31;
    const int gid  = blockIdx.x * 8 + wave;       // 0 .. B*H*T-1 (exact multiple)
    const int b    = gid / (Hc * Tc);
    const int rem  = gid - b * Hc * Tc;
    const int h    = rem / Tc;
    const int t    = rem - h * Tc;
    const float scale = 0.125f;                   // 1/sqrt(64)

    const float* qrow = q + ((size_t)(b * Tc + t)) * Dc + h * HDc;
    qs[wave][2 * lane]     = qrow[2 * lane];
    qs[wave][2 * lane + 1] = qrow[2 * lane + 1];
    __syncthreads();

    // scores: lane handles j, j+32 (lane 0 also j=64)
    for (int j = lane; j < NKEY; j += 32) {
        int col; bool valid;
        if (j < Kc) {
            col = t - WINc + j;
            valid = (col >= 0) && (col < Tc);
            if (!valid) col = 0;
        } else {
            col = gidx[j - Kc];
            valid = true;
        }
        const float* krow = k + ((size_t)(b * Tc + col)) * Dc + h * HDc;
        float s = 0.f;
        #pragma unroll 8
        for (int d = 0; d < HDc; ++d) s += qs[wave][d] * krow[d];
        as[wave][j] = valid ? s * scale : -INFINITY;
    }
    __syncthreads();

    // softmax over 65 entries
    float m = -INFINITY;
    for (int j = lane; j < NKEY; j += 32) m = fmaxf(m, as[wave][j]);
    #pragma unroll
    for (int off = 16; off > 0; off >>= 1) m = fmaxf(m, __shfl_xor(m, off));

    float ssum = 0.f;
    for (int j = lane; j < NKEY; j += 32) {
        float e = __expf(as[wave][j] - m);
        as[wave][j] = e;
        ssum += e;
    }
    #pragma unroll
    for (int off = 16; off > 0; off >>= 1) ssum += __shfl_xor(ssum, off);
    const float inv = 1.0f / ssum;

    for (int j = lane; j < NKEY; j += 32) as[wave][j] *= inv;
    __syncthreads();

    // scatter into full_attn (global columns win over local window overlap)
    float* arow = full_attn + (((size_t)(b * Hc + h)) * Tc + t) * Tc;
    for (int j = lane; j < NKEY; j += 32) {
        const float a = as[wave][j];
        if (j < Kc) {
            int col = t - WINc + j;
            if (col >= 0 && col < Tc && mask[col] == 0) arow[col] = a;
        } else {
            arow[gidx[j - Kc]] = a;
        }
    }

    // ctx: each lane owns 2 dims
    const int d0 = 2 * lane;
    float c0 = 0.f, c1 = 0.f;
    for (int j = 0; j < NKEY; ++j) {
        const float a = as[wave][j];
        int col = (j < Kc) ? min(max(t - WINc + j, 0), Tc - 1) : gidx[j - Kc];
        const float* vrow = v + ((size_t)(b * Tc + col)) * Dc + h * HDc;
        c0 += a * vrow[d0];
        c1 += a * vrow[d0 + 1];
    }
    float* crow = ctx + ((size_t)(b * Tc + t)) * Dc + h * HDc;
    crow[d0]     = c0;
    crow[d0 + 1] = c1;
}

// ---------------- launch ----------------
extern "C" void kernel_launch(void* const* d_in, const int* in_sizes, int n_in,
                              void* d_out, int out_size, void* d_ws, size_t ws_size,
                              hipStream_t stream) {
    const float* x    = (const float*)d_in[0];
    const int*   mask = (const int*)  d_in[1];
    const float* Wq   = (const float*)d_in[2];
    const float* bq   = (const float*)d_in[3];
    const float* Wk   = (const float*)d_in[4];
    const float* bk   = (const float*)d_in[5];
    const float* Wv   = (const float*)d_in[6];
    const float* bv   = (const float*)d_in[7];
    const float* Wo   = (const float*)d_in[8];
    const float* bo   = (const float*)d_in[9];

    float* out       = (float*)d_out;                         // (B,T,D)
    float* full_attn = out + (size_t)Bc * Tc * Dc;            // (B,H,T,T)

    // workspace carve-up (256B aligned)
    auto align256 = [](size_t v) { return (v + 255) & ~(size_t)255; };
    char* ws = (char*)d_ws;
    size_t off = 0;
    uint16_t* x_bf  = (uint16_t*)(ws + off); off = align256(off + (size_t)Mc * Dc * 2);
    uint16_t* wq_bf = (uint16_t*)(ws + off); off = align256(off + (size_t)Dc * Dc * 2);
    uint16_t* wk_bf = (uint16_t*)(ws + off); off = align256(off + (size_t)Dc * Dc * 2);
    uint16_t* wv_bf = (uint16_t*)(ws + off); off = align256(off + (size_t)Dc * Dc * 2);
    uint16_t* wo_bf = (uint16_t*)(ws + off); off = align256(off + (size_t)Dc * Dc * 2);
    float* qbuf     = (float*)(ws + off);    off = align256(off + (size_t)Mc * Dc * 4);
    float* kbuf     = (float*)(ws + off);    off = align256(off + (size_t)Mc * Dc * 4);
    float* vbuf     = (float*)(ws + off);    off = align256(off + (size_t)Mc * Dc * 4);
    float* ctxbuf   = (float*)(ws + off);    off = align256(off + (size_t)Mc * Dc * 4);
    uint16_t* ctx_bf = (uint16_t*)(ws + off); off = align256(off + (size_t)Mc * Dc * 2);
    int* gidx       = (int*)(ws + off);      off = align256(off + Gc * 4);

    // zero the dense attention output (only ~65/row are written by the kernel)
    (void)hipMemsetAsync(full_attn, 0, (size_t)Bc * Hc * Tc * Tc * sizeof(float), stream);

    // bf16 conversions (weights transposed so GEMM B-tiles are row-contiguous)
    {
        const int nx = Mc * Dc, nw = Dc * Dc;
        f2bf_kernel<<<(nx + 255) / 256, 256, 0, stream>>>(x, x_bf, nx);
        f2bf_transpose_kernel<<<(nw + 255) / 256, 256, 0, stream>>>(Wq, wq_bf);
        f2bf_transpose_kernel<<<(nw + 255) / 256, 256, 0, stream>>>(Wk, wk_bf);
        f2bf_transpose_kernel<<<(nw + 255) / 256, 256, 0, stream>>>(Wv, wv_bf);
        f2bf_transpose_kernel<<<(nw + 255) / 256, 256, 0, stream>>>(Wo, wo_bf);
    }

    build_gidx_kernel<<<1, 32, 0, stream>>>(mask, gidx);

    // QKV projections (WMMA bf16, async LDS staging)
    dim3 gemmGrid(Mc / 128, Dc / 128);
    gemm_bf16_kernel<<<gemmGrid, 256, 0, stream>>>(x_bf, wq_bf, bq, qbuf);
    gemm_bf16_kernel<<<gemmGrid, 256, 0, stream>>>(x_bf, wk_bf, bk, kbuf);
    gemm_bf16_kernel<<<gemmGrid, 256, 0, stream>>>(x_bf, wv_bf, bv, vbuf);

    // attention: one wave per (b,h,t) row
    attn_kernel<<<(Bc * Hc * Tc) / 8, 256, 0, stream>>>(qbuf, kbuf, vbuf, mask, gidx,
                                                        ctxbuf, full_attn);

    // output projection
    f2bf_kernel<<<(Mc * Dc + 255) / 256, 256, 0, stream>>>(ctxbuf, ctx_bf, Mc * Dc);
    gemm_bf16_kernel<<<gemmGrid, 256, 0, stream>>>(ctx_bf, wo_bf, bo, out);
}
